// ImpalaGNN_5952824673213
// MI455X (gfx1250) — matrix operand order repaired and checked by reading, not compiled, not run on hardware
//
#include <hip/hip_runtime.h>
#include <hip/hip_bf16.h>
#include <hip/hip_fp16.h>

// ---------------------------------------------------------------------------
// MI455X (gfx1250) implementation. All GEMM-shaped math (convs via implicit
// im2col, MHA projections, attention, FC) runs on v_wmma_f32_16x16x32_f16.
// wave32; block = 128 threads = 4 waves. Double-buffered LDS staging with a
// single barrier per K-chunk; GEMM waves compute 16x32 (2 WMMA, shared A frag).
// ---------------------------------------------------------------------------

typedef __attribute__((ext_vector_type(16))) _Float16 v16h;
typedef __attribute__((ext_vector_type(8)))  float    v8f;

#define LDT_PAD 34  // halfs per LDS tile row (32 data + pad)

// A fragment: 16x32 f16, row-major LDS tile. Per ISA 7.12.2:
// lane L: M=L&15, hi=L>>4; e[0..7]=A[M][8*hi..], e[8..15]=A[M][16+8*hi..].
__device__ __forceinline__ v16h frag_a(const _Float16* As, int lane) {
  const int m = lane & 15, hi = lane >> 4;
  const _Float16* p0 = As + m * LDT_PAD + 8 * hi;
  const _Float16* p1 = As + m * LDT_PAD + 16 + 8 * hi;
  v16h a;
#pragma unroll
  for (int j = 0; j < 8; ++j) { a[j] = p0[j]; a[8 + j] = p1[j]; }
  return a;
}

// B fragment: 32x16 f16 (KxN). LDS tile stored N-major (Bt[n][k]) so
// lane L: N=L&15, hi=L>>4; e[0..7]=B[8*hi..][N], e[8..15]=B[16+8*hi..][N].
__device__ __forceinline__ v16h frag_b(const _Float16* Bs, int lane) {
  const int n = lane & 15, hi = lane >> 4;
  const _Float16* p = Bs + n * LDT_PAD;
  v16h b;
#pragma unroll
  for (int j = 0; j < 8; ++j) { b[j] = p[8 * hi + j]; b[8 + j] = p[16 + 8 * hi + j]; }
  return b;
}

// ---------------------------------------------------------------------------
// Implicit-GEMM 3x3 conv, pad=1, stride=1. M = H*W, N = Cout, K = Cin*9.
// Each staging thread owns one fixed K column (128 % 32 == 0), so the
// (ci,ky,kx) decode happens once per chunk; spatial coords advance by +4.
// ---------------------------------------------------------------------------
__global__ __launch_bounds__(128) void conv3x3_wmma(
    const float* __restrict__ x, const float* __restrict__ w,
    const float* __restrict__ bias, float* __restrict__ y,
    int Cin, int H, int W, int Cout, int relu_in, float scale_in) {
  __shared__ _Float16 As[2][64 * LDT_PAD];
  __shared__ _Float16 Bs[2][16 * LDT_PAD];
  const int tid = threadIdx.x, lane = tid & 31, wv = tid >> 5;
  const int HW = H * W, K = Cin * 9;
  const int m_base = blockIdx.x * 64, n_base = blockIdx.y * 16;
  const int b = blockIdx.z;
  const int kcol = tid & 31, row0 = tid >> 5;

  auto stage = [&](int p, int k0) {
    // ---- A tile: im2col gather, fixed k column, rows row0+4j ----
    {
      const int kg = k0 + kcol;
      const bool kok = kg < K;
      int ci = 0, ky = 0, kx = 0;
      if (kok) { ci = kg / 9; const int r = kg - ci * 9; ky = r / 3; kx = r - ky * 3; }
      const float* xc = x + ((long long)b * Cin + ci) * HW;
      int mg = m_base + row0;
      int oy = mg / W, ox = mg - oy * W;
      _Float16* dst = &As[p][row0 * LDT_PAD + kcol];
#pragma unroll
      for (int j = 0; j < 16; ++j) {
        float v = 0.f;
        if (kok && mg < HW) {
          const int iy = oy + ky - 1, ix = ox + kx - 1;
          if (iy >= 0 && iy < H && ix >= 0 && ix < W) {
            v = xc[iy * W + ix];
            if (relu_in) v = v > 0.f ? v : 0.f;
            v *= scale_in;
          }
        }
        dst[j * 4 * LDT_PAD] = (_Float16)v;
        mg += 4; ox += 4; if (ox >= W) { ox -= W; ++oy; }
      }
    }
    // ---- B tile: OIHW weights, contiguous over K ----
    {
      const int kg = k0 + kcol;
#pragma unroll
      for (int j = 0; j < 4; ++j) {
        const int n = row0 + 4 * j, co = n_base + n;
        const float v = (co < Cout && kg < K) ? w[(long long)co * K + kg] : 0.f;
        Bs[p][n * LDT_PAD + kcol] = (_Float16)v;
      }
    }
  };

  v8f acc = {};
  stage(0, 0);
  __syncthreads();
  int par = 0;
  for (int k0 = 0; k0 < K; k0 += 32) {
    if (k0 + 32 < K) stage(par ^ 1, k0 + 32);   // prefill other buffer
    const v16h a  = frag_a(&As[par][(wv * 16) * LDT_PAD], lane);
    const v16h bf = frag_b(&Bs[par][0], lane);
    acc = __builtin_amdgcn_wmma_f32_16x16x32_f16(false, a, false, bf,
                                                 (short)0, acc, false, false);
    __syncthreads();
    par ^= 1;
  }

  const int n = lane & 15, hi = lane >> 4;
  const int co = n_base + n;
  if (co < Cout) {
    const float bv = bias ? bias[co] : 0.f;
#pragma unroll
    for (int r = 0; r < 8; ++r) {
      const int mg = m_base + wv * 16 + 8 * hi + r;   // M = r + 8*hi
      if (mg < HW)
        y[((long long)b * Cout + co) * HW + mg] = acc[r] + bv;
    }
  }
}

// ---------------------------------------------------------------------------
// Generic batched WMMA GEMM: C = act(alpha*(A @ Bop) + bias + addsrc).
// Block tile 64x32; each wave: 16 rows x 32 cols = 2 WMMA sharing one A frag.
// A: MxK row-major (lda). B: KxN (ldb) or NxK if transB. Batch index z
// decomposes as (z/nh, z%nh) with separate strides (attention head views).
// ---------------------------------------------------------------------------
__global__ __launch_bounds__(128) void gemm_wmma(
    const float* __restrict__ A, const float* __restrict__ B,
    const float* __restrict__ bias, const float* __restrict__ addsrc,
    float* __restrict__ C, int M, int N, int K, int lda, int ldb, int ldc,
    int nh, long long sA, long long sA2, long long sB, long long sB2,
    long long sC, long long sC2, int transB, float alpha, int relu_out) {
  __shared__ _Float16 As[2][64 * LDT_PAD];
  __shared__ _Float16 Bs[2][32 * LDT_PAD];
  const int tid = threadIdx.x, lane = tid & 31, wv = tid >> 5;
  const int m_base = blockIdx.x * 64, n_base = blockIdx.y * 32;
  const int z = blockIdx.z, zb = z / nh, zh = z % nh;
  const float* Ab = A + (long long)zb * sA + (long long)zh * sA2;
  const float* Bb = B + (long long)zb * sB + (long long)zh * sB2;
  const long long cbase = (long long)zb * sC + (long long)zh * sC2;
  float* Cb = C + cbase;
  const float* Rb = addsrc ? addsrc + cbase : nullptr;
  const int kcol = tid & 31, row0 = tid >> 5;

  auto stage = [&](int p, int k0) {
    const int kg = k0 + kcol;
    const bool kok = kg < K;
    {
      _Float16* dst = &As[p][row0 * LDT_PAD + kcol];
#pragma unroll
      for (int j = 0; j < 16; ++j) {
        const int mg = m_base + row0 + 4 * j;
        dst[j * 4 * LDT_PAD] =
            (_Float16)((kok && mg < M) ? Ab[(long long)mg * lda + kg] : 0.f);
      }
    }
    {
#pragma unroll
      for (int j = 0; j < 8; ++j) {
        const int n = row0 + 4 * j, ng = n_base + n;
        float v = 0.f;
        if (kok && ng < N)
          v = transB ? Bb[(long long)ng * ldb + kg] : Bb[(long long)kg * ldb + ng];
        Bs[p][n * LDT_PAD + kcol] = (_Float16)v;
      }
    }
  };

  v8f acc0 = {}, acc1 = {};
  stage(0, 0);
  __syncthreads();
  int par = 0;
  for (int k0 = 0; k0 < K; k0 += 32) {
    if (k0 + 32 < K) {
      // speculative prefetch of next chunks (lowers to global_prefetch_b8)
      if (m_base < M) __builtin_prefetch(&Ab[(long long)m_base * lda + k0 + 32], 0, 1);
      __builtin_prefetch(&Bb[(long long)(transB ? n_base * ldb : (long long)(k0 + 32) * ldb)], 0, 1);
      stage(par ^ 1, k0 + 32);
    }
    const v16h a  = frag_a(&As[par][(wv * 16) * LDT_PAD], lane);
    const v16h b0 = frag_b(&Bs[par][0], lane);
    const v16h b1 = frag_b(&Bs[par][16 * LDT_PAD], lane);
    acc0 = __builtin_amdgcn_wmma_f32_16x16x32_f16(false, a, false, b0,
                                                  (short)0, acc0, false, false);
    acc1 = __builtin_amdgcn_wmma_f32_16x16x32_f16(false, a, false, b1,
                                                  (short)0, acc1, false, false);
    __syncthreads();
    par ^= 1;
  }

  const int n = lane & 15, hi = lane >> 4;
  const float* biasp = bias;
#pragma unroll
  for (int t = 0; t < 2; ++t) {
    const int ng = n_base + 16 * t + n;
    if (ng < N) {
      const float bv = biasp ? biasp[ng] : 0.f;
      const v8f& acc = t ? acc1 : acc0;
#pragma unroll
      for (int r = 0; r < 8; ++r) {
        const int mg = m_base + wv * 16 + 8 * hi + r;
        if (mg < M) {
          float v = alpha * acc[r] + bv;
          if (Rb) v += Rb[(long long)mg * ldc + ng];
          if (relu_out) v = v > 0.f ? v : 0.f;
          Cb[(long long)mg * ldc + ng] = v;
        }
      }
    }
  }
}

// ------------------------- elementwise / reduction kernels ------------------
__global__ __launch_bounds__(256) void layernorm_k(
    const float* __restrict__ x, float* __restrict__ y,
    const float* __restrict__ g, const float* __restrict__ b, int L) {
  __shared__ float red[256];
  const long long row = blockIdx.x;
  const float* xr = x + row * L;
  float s = 0.f;
  for (int i = threadIdx.x; i < L; i += 256) s += xr[i];
  red[threadIdx.x] = s; __syncthreads();
  for (int o = 128; o > 0; o >>= 1) {
    if (threadIdx.x < o) red[threadIdx.x] += red[threadIdx.x + o];
    __syncthreads();
  }
  const float mean = red[0] / L; __syncthreads();
  float s2 = 0.f;
  for (int i = threadIdx.x; i < L; i += 256) { float d = xr[i] - mean; s2 += d * d; }
  red[threadIdx.x] = s2; __syncthreads();
  for (int o = 128; o > 0; o >>= 1) {
    if (threadIdx.x < o) red[threadIdx.x] += red[threadIdx.x + o];
    __syncthreads();
  }
  const float rstd = rsqrtf(red[0] / L + 1e-6f);
  float* yr = y + row * L;
  for (int i = threadIdx.x; i < L; i += 256)
    yr[i] = (xr[i] - mean) * rstd * g[i] + b[i];
}

__global__ __launch_bounds__(128) void softmax_k(float* __restrict__ x, int L) {
  __shared__ float red[128];
  float* xr = x + (long long)blockIdx.x * L;
  float mx = -3.4e38f;
  for (int i = threadIdx.x; i < L; i += 128) mx = fmaxf(mx, xr[i]);
  red[threadIdx.x] = mx; __syncthreads();
  for (int o = 64; o > 0; o >>= 1) {
    if (threadIdx.x < o) red[threadIdx.x] = fmaxf(red[threadIdx.x], red[threadIdx.x + o]);
    __syncthreads();
  }
  mx = red[0]; __syncthreads();
  float s = 0.f;
  for (int i = threadIdx.x; i < L; i += 128) { float e = __expf(xr[i] - mx); xr[i] = e; s += e; }
  red[threadIdx.x] = s; __syncthreads();
  for (int o = 64; o > 0; o >>= 1) {
    if (threadIdx.x < o) red[threadIdx.x] += red[threadIdx.x + o];
    __syncthreads();
  }
  const float inv = 1.f / red[0];
  for (int i = threadIdx.x; i < L; i += 128) xr[i] *= inv;
}

__global__ void maxpool_k(const float* __restrict__ x, float* __restrict__ y,
                          int C, int H, int W, int Ho, int Wo, long long total) {
  const long long idx = (long long)blockIdx.x * blockDim.x + threadIdx.x;
  if (idx >= total) return;
  const int wo = (int)(idx % Wo);
  const int ho = (int)((idx / Wo) % Ho);
  const int c  = (int)((idx / ((long long)Wo * Ho)) % C);
  const int b  = (int)(idx / ((long long)Wo * Ho * C));
  float m = -3.4e38f;
#pragma unroll
  for (int ky = 0; ky < 3; ++ky)
#pragma unroll
    for (int kx = 0; kx < 3; ++kx) {
      const int iy = ho * 2 - 1 + ky, ix = wo * 2 - 1 + kx;
      if (iy >= 0 && iy < H && ix >= 0 && ix < W)
        m = fmaxf(m, x[(((long long)b * C + c) * H + iy) * W + ix]);
    }
  y[idx] = m;
}

__global__ void relu_k(float* x, long long n) {
  const long long i = (long long)blockIdx.x * blockDim.x + threadIdx.x;
  if (i < n) x[i] = x[i] > 0.f ? x[i] : 0.f;
}
__global__ void add_k(float* x, const float* y, long long n) {
  const long long i = (long long)blockIdx.x * blockDim.x + threadIdx.x;
  if (i < n) x[i] += y[i];
}

// (B,30,11,11) -> (B,121,32): relu + normalized coord channels 30/31.
__global__ void build_e_k(const float* __restrict__ x, float* __restrict__ e, int Bn) {
  const long long idx = (long long)blockIdx.x * blockDim.x + threadIdx.x;
  const long long total = (long long)Bn * 121 * 32;
  if (idx >= total) return;
  const int c = (int)(idx & 31);
  const long long t = idx >> 5;
  const int n = (int)(t % 121), b = (int)(t / 121);
  float v;
  if (c < 30) { v = x[((long long)b * 30 + c) * 121 + n]; v = v > 0.f ? v : 0.f; }
  else if (c == 30) v = (float)(n / 11) / 11.f;
  else              v = (float)(n % 11) / 11.f;
  e[idx] = v;
}

// flat[b][c*121+n] = relu(e2[b][n][c] + eres[b][n][c])  (channel-major flatten)
__global__ void repack_flat_k(const float* __restrict__ e2,
                              const float* __restrict__ eres,
                              float* __restrict__ flat, int Bn) {
  const long long idx = (long long)blockIdx.x * blockDim.x + threadIdx.x;
  const long long total = (long long)Bn * 121 * 32;
  if (idx >= total) return;
  const int c = (int)(idx & 31);
  const long long t = idx >> 5;
  const int n = (int)(t % 121), b = (int)(t / 121);
  float v = e2[idx] + eres[idx];
  v = v > 0.f ? v : 0.f;
  flat[((long long)b * 32 + c) * 121 + n] = v;
}

// ---------------------------------------------------------------------------
extern "C" void kernel_launch(void* const* d_in, const int* in_sizes, int n_in,
                              void* d_out, int out_size, void* d_ws, size_t ws_size,
                              hipStream_t stream) {
  (void)in_sizes; (void)n_in; (void)out_size; (void)ws_size;
  const int B = 256, NTOK = 121;
  const long long DROW = (long long)B * NTOK;  // 30976
  const float* inp = (const float*)d_in[0];
  const float* rnn = (const float*)d_in[1];
  auto P = [&](int i) { return (const float*)d_in[i]; };

  // params flattened in setup_inputs() insertion order
  const float *c0w = P(3), *c0b = P(4);
  struct RC { const float *w1, *b1, *w2, *b2; };
  RC rc0a{P(5), P(6), P(7), P(8)},    rc0b{P(9), P(10), P(11), P(12)};
  const float *c1w = P(13), *c1b = P(14);
  RC rc1a{P(15), P(16), P(17), P(18)}, rc1b{P(19), P(20), P(21), P(22)};
  const float *c2w = P(23), *c2b = P(24);
  struct MHAP { const float *wq, *wk, *wv, *wo, *ge, *be, *g, *b; };
  auto mp = [&](int o) { return MHAP{P(o), P(o+1), P(o+2), P(o+3), P(o+4), P(o+5), P(o+6), P(o+7)}; };
  MHAP m0a = mp(25), m0b = mp(33), m1a = mp(41), m1b = mp(49);
  const float *fcw = P(57), *fcb = P(58), *crw = P(59), *crb = P(60);

  // workspace arena (conv region reused by MHA Q/K/V/SC/NV); ~580 MB total
  char* ws = (char*)d_ws;
  float* X84 = (float*)(ws + 0);            // 256*16*84*84*4 = 115605504
  float* T1  = (float*)(ws + 115605504);    // 57802752
  float* T2  = (float*)(ws + 173408256);    // 57802752
  float* T3  = (float*)(ws + 231211008);    // 57802752
  float* Q   = (float*)(ws + 0);            // 126877696
  float* Kb  = (float*)(ws + 126877696);
  float* V   = (float*)(ws + 253755392);
  float* SC  = (float*)(ws + 380633088);    // 59969536
  float* NV  = (float*)(ws + 440602624);    // 126877696
  float* ER  = (float*)(ws + 567480320);    // 3964928
  float* EB  = (float*)(ws + 571445248);    // 3964928
  float* MO  = (float*)(ws + 575410176);    // 3964928
  float* FLAT = Q;                          // reuse (Q dead by then)
  const long long NE = DROW * 32;
  const size_t E_BYTES = (size_t)NE * sizeof(float);

  auto conv = [&](const float* x, const float* w, const float* bias, float* y,
                  int Cin, int Hd, int Cout, int relu_in, float scale) {
    dim3 g((Hd * Hd + 63) / 64, (Cout + 15) / 16, B);
    conv3x3_wmma<<<g, 128, 0, stream>>>(x, w, bias, y, Cin, Hd, Hd, Cout, relu_in, scale);
  };
  auto pool = [&](const float* x, float* y, int C, int H, int Ho) {
    long long total = (long long)B * C * Ho * Ho;
    maxpool_k<<<(unsigned)((total + 255) / 256), 256, 0, stream>>>(x, y, C, H, H, Ho, Ho, total);
  };
  auto relu = [&](float* x, long long n) {
    relu_k<<<(unsigned)((n + 255) / 256), 256, 0, stream>>>(x, n);
  };
  auto add = [&](float* x, const float* y, long long n) {
    add_k<<<(unsigned)((n + 255) / 256), 256, 0, stream>>>(x, y, n);
  };
  auto gemm = [&](const float* A, const float* Bm, const float* bias, const float* adds,
                  float* C, int M, int N, int K, int lda, int ldb, int ldc,
                  int nh, long long sA, long long sA2, long long sB, long long sB2,
                  long long sC, long long sC2, int transB, float alpha, int relu_out,
                  int batch) {
    dim3 g((M + 63) / 64, (N + 31) / 32, batch);
    gemm_wmma<<<g, 128, 0, stream>>>(A, Bm, bias, adds, C, M, N, K, lda, ldb, ldc,
                                     nh, sA, sA2, sB, sB2, sC, sC2, transB, alpha, relu_out);
  };
  auto res_conv = [&](float* xb, float* t1, float* t2, const RC& p, int C, int Hd) {
    conv(xb, p.w1, p.b1, t1, C, Hd, C, 1, 1.f);
    conv(t1, p.w2, p.b2, t2, C, Hd, C, 1, 1.f);
    add(xb, t2, (long long)B * C * Hd * Hd);
  };
  auto run_mha = [&](const float* ein, float* eout, const MHAP& p) {
    const long long HS = (long long)NTOK * 1024;        // per-image q/k/v stride
    // projections (M=30976,K=32,N=1024) + LN over 1024
    gemm(ein, p.wq, nullptr, nullptr, Q,  (int)DROW, 1024, 32, 32, 1024, 1024,
         1, 0, 0, 0, 0, 0, 0, 0, 1.f, 0, 1);
    gemm(ein, p.wk, nullptr, nullptr, Kb, (int)DROW, 1024, 32, 32, 1024, 1024,
         1, 0, 0, 0, 0, 0, 0, 0, 1.f, 0, 1);
    gemm(ein, p.wv, nullptr, nullptr, V,  (int)DROW, 1024, 32, 32, 1024, 1024,
         1, 0, 0, 0, 0, 0, 0, 0, 1.f, 0, 1);
    layernorm_k<<<(unsigned)DROW, 256, 0, stream>>>(Q,  Q,  p.ge, p.be, 1024);
    layernorm_k<<<(unsigned)DROW, 256, 0, stream>>>(Kb, Kb, p.ge, p.be, 1024);
    layernorm_k<<<(unsigned)DROW, 256, 0, stream>>>(V,  V,  p.ge, p.be, 1024);
    // scores = (Q @ K^T) / 16 per (b,h): M=N=121, K=256
    gemm(Q, Kb, nullptr, nullptr, SC, 121, 121, 256, 1024, 1024, 121,
         4, HS, 256, HS, 256, (long long)4 * 121 * 121, (long long)121 * 121,
         1, 0.0625f, 0, B * 4);
    softmax_k<<<(unsigned)(B * 4 * 121), 128, 0, stream>>>(SC, 121);
    // nv = W @ V per (b,h): M=121, K=121, N=256
    gemm(SC, V, nullptr, nullptr, NV, 121, 256, 121, 121, 1024, 1024,
         4, (long long)4 * 121 * 121, (long long)121 * 121, HS, 256, HS, 256,
         0, 1.f, 0, B * 4);
    // out = nv @ wo + residual(ein), then LN over 32
    gemm(NV, p.wo, nullptr, ein, MO, (int)DROW, 32, 1024, 1024, 32, 32,
         1, 0, 0, 0, 0, 0, 0, 0, 1.f, 0, 1);
    layernorm_k<<<(unsigned)DROW, 256, 0, stream>>>(MO, eout, p.g, p.b, 32);
  };

  // ---- conv tower ----
  conv(inp, c0w, c0b, X84, 3, 84, 16, 0, 1.f / 255.f);
  pool(X84, T1, 16, 84, 42);
  res_conv(T1, T2, T3, rc0a, 16, 42);
  res_conv(T1, T2, T3, rc0b, 16, 42);
  conv(T1, c1w, c1b, X84, 16, 42, 32, 0, 1.f);
  pool(X84, T1, 32, 42, 21);
  res_conv(T1, T2, T3, rc1a, 32, 21);
  res_conv(T1, T2, T3, rc1b, 32, 21);
  conv(T1, c2w, c2b, T2, 32, 21, 30, 0, 1.f);
  pool(T2, T3, 30, 21, 11);

  // ---- relational MHA head ----
  build_e_k<<<(unsigned)((NE + 255) / 256), 256, 0, stream>>>(T3, ER, B);
  run_mha(ER, EB, m0a);   relu(EB, NE);
  run_mha(EB, EB, m0b);   relu(EB, NE);
  add(EB, ER, NE);        // e2 + e
  relu(EB, NE);           // relu at entry of second res_mha
  hipMemcpyAsync(ER, EB, E_BYTES, hipMemcpyDeviceToDevice, stream);
  run_mha(EB, EB, m1a);   relu(EB, NE);
  run_mha(EB, EB, m1b);   relu(EB, NE);
  repack_flat_k<<<(unsigned)((NE + 255) / 256), 256, 0, stream>>>(EB, ER, FLAT, B);

  // ---- FC head: out = relu(flat @ fcw + fcb); value = out @ crw + crb ----
  float* OUT = (float*)d_out + 256;
  float* VAL = (float*)d_out;
  gemm(FLAT, fcw, fcb, nullptr, OUT, 256, 256, 3872, 3872, 256, 256,
       1, 0, 0, 0, 0, 0, 0, 0, 1.f, 1, 1);
  gemm(OUT, crw, crb, nullptr, VAL, 256, 1, 256, 256, 1, 1,
       1, 0, 0, 0, 0, 0, 0, 0, 1.f, 0, 1);
  hipMemcpyAsync((float*)d_out + 256 + 65536, rnn, 256 * sizeof(float),
                 hipMemcpyDeviceToDevice, stream);
}
